// BiLSTM_80848464379960
// MI455X (gfx1250) — matrix operand order repaired and checked by reading, not compiled, not run on hardware
//
#include <hip/hip_runtime.h>
#include <hip/hip_bf16.h>

// ---------------- Types for WMMA (gfx1250, wave32) ----------------
typedef __attribute__((ext_vector_type(8)))  float          v8f;
typedef __attribute__((ext_vector_type(16))) __bf16         v16bf;
typedef __attribute__((ext_vector_type(8)))  unsigned short us8v;

union Frag16 {
    us8v  u[2];   // two 16-byte chunks
    v16bf bf;
};

__device__ __forceinline__ unsigned short f2bf_(float x) {
    union { float f; unsigned u; } v; v.f = x;
    unsigned r = v.u + 0x7FFFu + ((v.u >> 16) & 1u);   // RNE
    return (unsigned short)(r >> 16);
}
__device__ __forceinline__ float sigmoidf_(float x) { return 1.0f / (1.0f + __expf(-x)); }

__device__ __forceinline__ v8f wmma_bf16(Frag16 a, Frag16 b, v8f c) {
    return __builtin_amdgcn_wmma_f32_16x16x32_bf16(false, a.bf, false, b.bf, (short)0, c, false, false);
}

// Problem constants
#define BB   32
#define TT   512
#define EMB_ 300
#define UU   128
#define NC   279

// ---------------- Weight convert: fp32 [K][N] -> bf16 [Npad][Kpad] (transposed, zero-padded) ----
__global__ void k_convT(const float* __restrict__ W, unsigned short* __restrict__ out,
                        int K, int N, int Kpad, int Npad) {
    int idx = blockIdx.x * blockDim.x + threadIdx.x;
    int total = Npad * Kpad;
    if (idx >= total) return;
    int n = idx / Kpad, k = idx % Kpad;
    out[idx] = (n < N && k < K) ? f2bf_(W[(size_t)k * N + n]) : (unsigned short)0;
}

// ---------------- fp32 -> bf16 elementwise ----------------
__global__ void k_f2bf(const float* __restrict__ in, unsigned short* __restrict__ out, int n) {
    int i = blockIdx.x * blockDim.x + threadIdx.x;
    if (i < n) out[i] = f2bf_(in[i]);
}

// ---------------- Embedding gather -> bf16 [16384][320] (cols 300..319 zero) -------------
__global__ void k_embed(const int* __restrict__ ids, const float* __restrict__ table,
                        unsigned short* __restrict__ xbf) {
    int row = blockIdx.x;                 // b*T + t
    int id  = ids[row];
    const float* src = table + (size_t)id * EMB_;
    unsigned short* dst = xbf + (size_t)row * 320;
    for (int j = threadIdx.x; j < 320; j += blockDim.x)
        dst[j] = (j < EMB_) ? f2bf_(src[j]) : (unsigned short)0;
}

// ---------------- Generic WMMA GEMM: C[M,N] = A[M,Kpad](bf16) * BT[Npad,Kpad](bf16)^T + bias ----
// block = 256 threads (8 waves). Block tile: 128(M) x 64(N). Wave: 16(M) x 64(N).
// permute!=0: output row b*512+t is written at row t*32+b (z layout [T][B][N]).
__global__ void k_gemm(const unsigned short* __restrict__ A, const unsigned short* __restrict__ BT,
                       const float* __restrict__ bias, float* __restrict__ C,
                       int Kpad, int N, int ldc, int permute) {
    const int tid  = threadIdx.x;
    const int wave = tid >> 5;
    const int lane = tid & 31;
    const int l16  = lane & 15;
    const int half = lane >> 4;

    const int mrow  = blockIdx.y * 128 + wave * 16;
    const int nbase = blockIdx.x * 64;

    v8f acc[4] = {};
    const int ktiles = Kpad >> 5;

    for (int k = 0; k < ktiles; ++k) {
        // A fragment (16x32 bf16): lane half 0 -> K 0..15 ; half 1 -> K 16..31
        Frag16 a;
        const unsigned short* ap = A + (size_t)(mrow + l16) * Kpad + k * 32 + half * 16;
        a.u[0] = *(const us8v*)ap;
        a.u[1] = *(const us8v*)(ap + 8);
#pragma unroll
        for (int nt = 0; nt < 4; ++nt) {
            // B fragment (32x16 bf16, BT is [N][Kpad]): lane holds 16 contiguous K at half*16
            Frag16 b;
            const unsigned short* bp = BT + (size_t)(nbase + nt * 16 + l16) * Kpad + k * 32 + half * 16;
            b.u[0] = *(const us8v*)bp;
            b.u[1] = *(const us8v*)(bp + 8);
            acc[nt] = wmma_bf16(a, b, acc[nt]);
        }
    }
    // Epilogue. C/D layout: VGPR r, lanes<16 -> M=r, lanes>=16 -> M=r+8, N=l16.
#pragma unroll
    for (int nt = 0; nt < 4; ++nt) {
        int col = nbase + nt * 16 + l16;
        if (col >= N) continue;
        float bv = bias[col];
#pragma unroll
        for (int r = 0; r < 8; ++r) {
            int row  = mrow + r + half * 8;
            int orow = permute ? ((row & 511) * 32 + (row >> 9)) : row;
            C[(size_t)orow * ldc + col] = acc[nt][r] + bv;
        }
    }
}

// ---------------- LSTM recurrence: one workgroup per direction -------------------
// z layout: [T][32][512] fp32 (gates i,f,g,o in cols 0..127,128..255,256..383,384..511)
// WrT: [512][128] bf16 (transposed recurrent weights). enc: [B][T][256] fp32.
// Wave w (of 16): m = w&1 (batch tile), ut = w>>1 (unit tile 0..7). Its 4 gate tiles are
// columns ut*16 + g*128, so i/f/g/o for its (b,u) cells stay in its own registers; c and
// previous h live in VGPRs across all 512 steps. Its 16 B-fragments of WrT are loop-
// invariant and stay resident in VGPRs for the whole scan. Only the masked h round-trips
// through 8 KB of LDS (bf16) between steps.
__global__ void k_lstm(const float* __restrict__ zf, const float* __restrict__ zb,
                       const unsigned short* __restrict__ wrfT, const unsigned short* __restrict__ wrbT,
                       const int* __restrict__ ids, float* __restrict__ enc, int residual) {
    const int dir = blockIdx.x;
    const float*          z   = dir ? zb   : zf;
    const unsigned short* WrT = dir ? wrbT : wrfT;

    __shared__ unsigned short hbf[BB * UU];   // h as bf16, [b][u], 8 KB

    const int tid  = threadIdx.x;
    const int wave = tid >> 5;
    const int lane = tid & 31;
    const int l16  = lane & 15;
    const int half = lane >> 4;
    const int m    = wave & 1;
    const int ut   = wave >> 1;

    const int brow0 = m * 16 + half * 8;      // + r gives batch index b
    const int ucol  = ut * 16 + l16;          // unit index owned by this lane column

    // Resident recurrent-weight fragments: [gate][ktile]
    Frag16 bfr[4][4];
#pragma unroll
    for (int g = 0; g < 4; ++g)
#pragma unroll
        for (int k = 0; k < 4; ++k) {
            const unsigned short* bp = &WrT[(size_t)(ut * 16 + g * 128 + l16) * UU + k * 32 + half * 16];
            bfr[g][k].u[0] = *(const us8v*)bp;
            bfr[g][k].u[1] = *(const us8v*)(bp + 8);
        }

    v8f c_st = {};
    v8f h_pv = {};

    for (int i = tid; i < BB * UU; i += blockDim.x) hbf[i] = 0;
    __syncthreads();

    for (int s = 0; s < TT; ++s) {
        const int t = dir ? (TT - 1 - s) : s;

        // z accumulators depend only on t -> issue loads BEFORE the barrier so they
        // overlap with the barrier wait on the serial critical path.
        v8f acc[4];
        const float* zrow = z + (size_t)t * (BB * 512);
#pragma unroll
        for (int g = 0; g < 4; ++g) {
            int col = ut * 16 + g * 128 + l16;
#pragma unroll
            for (int r = 0; r < 8; ++r)
                acc[g][r] = zrow[(size_t)(brow0 + r) * 512 + col];
        }

        __syncthreads();   // h from previous step visible in hbf

        // A fragments of h (16x32 bf16), 4 k-tiles covering K=128
        Frag16 a[4];
#pragma unroll
        for (int k = 0; k < 4; ++k) {
            const unsigned short* p = &hbf[(m * 16 + l16) * UU + k * 32 + half * 16];
            a[k].u[0] = *(const us8v*)p;
            a[k].u[1] = *(const us8v*)(p + 8);
        }

        // z += h @ Wr  (16 WMMAs per wave, operands all in registers)
#pragma unroll
        for (int g = 0; g < 4; ++g)
#pragma unroll
            for (int k = 0; k < 4; ++k)
                acc[g] = wmma_bf16(a[k], bfr[g][k], acc[g]);

        __syncthreads();   // everyone done reading hbf before we overwrite it

        // gates + state update (all in registers for this wave's cells)
#pragma unroll
        for (int r = 0; r < 8; ++r) {
            float ig = sigmoidf_(acc[0][r]);
            float fg = sigmoidf_(acc[1][r]);
            float gg = tanhf(acc[2][r]);
            float og = sigmoidf_(acc[3][r]);
            float cn = fg * c_st[r] + ig * gg;
            float hn = og * tanhf(cn);
            int   b_ = brow0 + r;
            float mt = (ids[b_ * TT + t] != 0) ? 1.0f : 0.0f;   // per-lane mask, L1-broadcast
            c_st[r] = mt * cn + (1.0f - mt) * c_st[r];
            float h = mt * hn + (1.0f - mt) * h_pv[r];
            h_pv[r] = h;

            hbf[b_ * UU + ucol] = f2bf_(h);
            size_t eidx = ((size_t)b_ * TT + t) * 256 + dir * UU + ucol;
            if (residual) enc[eidx] += h; else enc[eidx] = h;
        }
    }
}

// ---------------- Softmax in place: one wave per row of 279 ----------------
__global__ void k_softmax(float* __restrict__ out, int rows, int cols) {
    int wave = (blockIdx.x * blockDim.x + threadIdx.x) >> 5;
    int lane = threadIdx.x & 31;
    if (wave >= rows) return;
    float* row = out + (size_t)wave * cols;

    float mx = -3.4e38f;
    for (int c = lane; c < cols; c += 32) mx = fmaxf(mx, row[c]);
    for (int o = 16; o > 0; o >>= 1) mx = fmaxf(mx, __shfl_xor(mx, o, 32));

    float sum = 0.0f;
    for (int c = lane; c < cols; c += 32) { float e = __expf(row[c] - mx); row[c] = e; sum += e; }
    for (int o = 16; o > 0; o >>= 1) sum += __shfl_xor(sum, o, 32);

    float inv = 1.0f / sum;
    for (int c = lane; c < cols; c += 32) row[c] *= inv;
}

// ---------------- Host side ----------------
extern "C" void kernel_launch(void* const* d_in, const int* in_sizes, int n_in,
                              void* d_out, int out_size, void* d_ws, size_t ws_size,
                              hipStream_t stream) {
    (void)in_sizes; (void)n_in; (void)out_size; (void)ws_size;
    const int*   ids   = (const int*)  d_in[0];
    const float* emb   = (const float*)d_in[1];
    const float* fw0_k = (const float*)d_in[2],  *fw0_r = (const float*)d_in[3],  *fw0_b = (const float*)d_in[4];
    const float* bw0_k = (const float*)d_in[5],  *bw0_r = (const float*)d_in[6],  *bw0_b = (const float*)d_in[7];
    const float* fw1_k = (const float*)d_in[8],  *fw1_r = (const float*)d_in[9],  *fw1_b = (const float*)d_in[10];
    const float* bw1_k = (const float*)d_in[11], *bw1_r = (const float*)d_in[12], *bw1_b = (const float*)d_in[13];
    const float* cls_w = (const float*)d_in[14], *cls_b = (const float*)d_in[15];
    float* out = (float*)d_out;

    char* ws = (char*)d_ws;
    size_t off = 0;
    auto alloc = [&](size_t bytes) { size_t r = off; off += (bytes + 255) & ~(size_t)255; return r; };

    const int M = BB * TT;                       // 16384
    unsigned short* wk0fT = (unsigned short*)(ws + alloc((size_t)512 * 320 * 2));
    unsigned short* wk0bT = (unsigned short*)(ws + alloc((size_t)512 * 320 * 2));
    unsigned short* wk1fT = (unsigned short*)(ws + alloc((size_t)512 * 256 * 2));
    unsigned short* wk1bT = (unsigned short*)(ws + alloc((size_t)512 * 256 * 2));
    unsigned short* wr0fT = (unsigned short*)(ws + alloc((size_t)512 * 128 * 2));
    unsigned short* wr0bT = (unsigned short*)(ws + alloc((size_t)512 * 128 * 2));
    unsigned short* wr1fT = (unsigned short*)(ws + alloc((size_t)512 * 128 * 2));
    unsigned short* wr1bT = (unsigned short*)(ws + alloc((size_t)512 * 128 * 2));
    unsigned short* clsT  = (unsigned short*)(ws + alloc((size_t)320 * 256 * 2));
    unsigned short* xbf   = (unsigned short*)(ws + alloc((size_t)M * 320 * 2));   // also reused as act bf16
    float*          encA  = (float*)         (ws + alloc((size_t)M * 256 * 4));
    float*          z_f   = (float*)         (ws + alloc((size_t)M * 512 * 4));
    float*          z_b   = (float*)         (ws + alloc((size_t)M * 512 * 4));

    auto convT = [&](const float* W, unsigned short* o, int K, int N, int Kp, int Np) {
        int total = Np * Kp;
        k_convT<<<(total + 255) / 256, 256, 0, stream>>>(W, o, K, N, Kp, Np);
    };
    convT(fw0_k, wk0fT, 300, 512, 320, 512);
    convT(bw0_k, wk0bT, 300, 512, 320, 512);
    convT(fw1_k, wk1fT, 256, 512, 256, 512);
    convT(bw1_k, wk1bT, 256, 512, 256, 512);
    convT(fw0_r, wr0fT, 128, 512, 128, 512);
    convT(bw0_r, wr0bT, 128, 512, 128, 512);
    convT(fw1_r, wr1fT, 128, 512, 128, 512);
    convT(bw1_r, wr1bT, 128, 512, 128, 512);
    convT(cls_w, clsT,  256, 279, 256, 320);

    // embedding gather -> bf16
    k_embed<<<M, 128, 0, stream>>>(ids, emb, xbf);

    // layer 0 input projections (permuted z layout [T][B][512])
    k_gemm<<<dim3(8, 128), 256, 0, stream>>>(xbf, wk0fT, fw0_b, z_f, 320, 512, 512, 1);
    k_gemm<<<dim3(8, 128), 256, 0, stream>>>(xbf, wk0bT, bw0_b, z_b, 320, 512, 512, 1);
    // layer 0 recurrence (fwd+bwd concurrently), enc0 = h
    k_lstm<<<2, 512, 0, stream>>>(z_f, z_b, wr0fT, wr0bT, ids, encA, 0);

    // enc0 -> bf16 (reuse xbf buffer)
    k_f2bf<<<(M * 256 + 255) / 256, 256, 0, stream>>>(encA, xbf, M * 256);

    // layer 1 input projections
    k_gemm<<<dim3(8, 128), 256, 0, stream>>>(xbf, wk1fT, fw1_b, z_f, 256, 512, 512, 1);
    k_gemm<<<dim3(8, 128), 256, 0, stream>>>(xbf, wk1bT, bw1_b, z_b, 256, 512, 512, 1);
    // layer 1 recurrence with fused residual: encA += h
    k_lstm<<<2, 512, 0, stream>>>(z_f, z_b, wr1fT, wr1bT, ids, encA, 1);

    // inner -> bf16 (reuse xbf), classifier GEMM straight into d_out, softmax in place
    k_f2bf<<<(M * 256 + 255) / 256, 256, 0, stream>>>(encA, xbf, M * 256);
    k_gemm<<<dim3(5, 128), 256, 0, stream>>>(xbf, clsT, cls_b, out, 256, NC, NC, 0);
    k_softmax<<<(M + 7) / 8, 256, 0, stream>>>(out, M, NC);
}
